// basic_block_68788196212855
// MI455X (gfx1250) — matrix operand order repaired and checked by reading, not compiled, not run on hardware
//
#include <hip/hip_runtime.h>

// ---------------------------------------------------------------------------
// Sizes fixed by the reference: B=8, N=256, d_in=256, d_mid=128, d_out=256
// ---------------------------------------------------------------------------
#define BB 8
#define NN 256
#define DIN 256
#define DMID 128
#define DOUT 256
#define TS 40   // LDS tile row stride in ushorts (80 B: 16-B aligned for b128 async)

typedef __attribute__((ext_vector_type(16))) __bf16 v16bf;
typedef __attribute__((ext_vector_type(8)))  float  v8f;

union BF16Frag { v16bf v; unsigned int u[8]; };
union F32Acc   { v8f v; float f[8]; };

static __device__ __forceinline__ unsigned short f2bf(float f) {
  unsigned u = __float_as_uint(f);
  unsigned r = u + 0x7FFFu + ((u >> 16) & 1u);   // RNE
  return (unsigned short)(r >> 16);
}

// |a - b| on packed 2 x bf16: v_pk_add_bf16 with src1 negated, then clear signs.
static __device__ __forceinline__ unsigned pk_absdiff_bf16(unsigned a, unsigned b) {
  unsigned d;
  asm("v_pk_add_bf16 %0, %1, %2 neg_lo:[0,1] neg_hi:[0,1]"
      : "=v"(d) : "v"(a), "v"(b));
  return d & 0x7fff7fffu;
}

// Async DMA: memory -> LDS, tracked by ASYNCcnt (CDNA5).
static __device__ __forceinline__ void async_b128(unsigned lds, unsigned voff,
                                                  unsigned long long saddr) {
  asm volatile("global_load_async_to_lds_b128 %0, %1, %2 offset:0"
               :: "v"(lds), "v"(voff), "s"(saddr) : "memory");
}
static __device__ __forceinline__ void async_b128x2(unsigned lds, unsigned voff,
                                                    unsigned long long saddr) {
  asm volatile("global_load_async_to_lds_b128 %0, %1, %2 offset:0"
               :: "v"(lds), "v"(voff), "s"(saddr) : "memory");
  asm volatile("global_load_async_to_lds_b128 %0, %1, %2 offset:16"
               :: "v"(lds), "v"(voff), "s"(saddr) : "memory");
}
static __device__ __forceinline__ void async_b128x4(unsigned lds, unsigned voff,
                                                    unsigned long long saddr) {
  asm volatile("global_load_async_to_lds_b128 %0, %1, %2 offset:0"
               :: "v"(lds), "v"(voff), "s"(saddr) : "memory");
  asm volatile("global_load_async_to_lds_b128 %0, %1, %2 offset:16"
               :: "v"(lds), "v"(voff), "s"(saddr) : "memory");
  asm volatile("global_load_async_to_lds_b128 %0, %1, %2 offset:32"
               :: "v"(lds), "v"(voff), "s"(saddr) : "memory");
  asm volatile("global_load_async_to_lds_b128 %0, %1, %2 offset:48"
               :: "v"(lds), "v"(voff), "s"(saddr) : "memory");
}
static __device__ __forceinline__ void wait_async0() {
  asm volatile("s_wait_asynccnt 0x0" ::: "memory");
}

// ---------------------------------------------------------------------------
// Core: per-(b,i) block GEMM  H[j,c] = sum_d |x[b,i,d]-x[b,j,d]| * W1[c,d]
// 256 threads = 8 waves; wave w owns j rows [jbase+16w, +16) x 128 channels.
// Double-buffered GLOBAL_LOAD_ASYNC_TO_LDS_B128 staging overlaps the DMA for
// chunk kc+1 with the WMMA chain of chunk kc.
// ---------------------------------------------------------------------------
__device__ __forceinline__ void h_gemm(const unsigned short* __restrict__ xg,
                                       const unsigned short* __restrict__ w1g,
                                       unsigned short* xjS,   // [2][128][TS]
                                       unsigned short* w1S,   // [2][128][TS] (c-major)
                                       const unsigned short* xiS, // [256]
                                       int b, int jbase, int tid, F32Acc* acc) {
  const int wave = tid >> 5, lane = tid & 31;
  const int lrow = lane & 15, half = lane >> 4;
#pragma unroll
  for (int ct = 0; ct < 8; ++ct)
#pragma unroll
    for (int r = 0; r < 8; ++r) acc[ct].f[r] = 0.f;

  const int row = tid >> 1, seg = tid & 1;
  const unsigned long long sx = (unsigned long long)(size_t)(xg + ((size_t)b << 16));
  const unsigned long long sw = (unsigned long long)(size_t)w1g;
  unsigned lds_x[2], lds_w[2];
#pragma unroll
  for (int p = 0; p < 2; ++p) {
    lds_x[p] = (unsigned)(size_t)(xjS + p * 128 * TS + row * TS + seg * 16);
    lds_w[p] = (unsigned)(size_t)(w1S + p * 128 * TS + row * TS + seg * 16);
  }
  auto issue = [&](int kc) {
    const int p = kc & 1, k0 = kc * 32;
    // x[b, jbase..jbase+127, k0..k0+31]  (32 B per lane)
    async_b128x2(lds_x[p], (unsigned)((((jbase + row) << 8) + k0 + seg * 16) * 2), sx);
    // W1[c=row, k0..k0+31]
    async_b128x2(lds_w[p], (unsigned)(((row << 8) + k0 + seg * 16) * 2), sw);
  };

  issue(0);
  for (int kc = 0; kc < 8; ++kc) {
    wait_async0();
    __syncthreads();                 // chunk kc ready; all readers of other buf done
    if (kc < 7) issue(kc + 1);       // overlap next DMA with this chunk's WMMAs
    const unsigned short* xjB = xjS + (kc & 1) * 128 * TS;
    const unsigned short* w1B = w1S + (kc & 1) * 128 * TS;
    const int k0 = kc * 32;

    // A fragment: 16x32 bf16, row M = lrow, K per ISA layout
    BF16Frag A;
    const unsigned short* xjrow = xjB + (wave * 16 + lrow) * TS;
#pragma unroll
    for (int vg = 0; vg < 8; ++vg) {
      int kl = ((vg & 4) ? 16 : 0) + half * 8 + 2 * (vg & 3);
      unsigned xjp = *reinterpret_cast<const unsigned*>(xjrow + kl);
      unsigned xip = *reinterpret_cast<const unsigned*>(xiS + k0 + kl);
      A.u[vg] = pk_absdiff_bf16(xip, xjp);
    }
    // B fragments: 32x16 bf16 per channel tile; col = lrow, K half by lane
#pragma unroll
    for (int ct = 0; ct < 8; ++ct) {
      BF16Frag Bf;
      const unsigned short* wrow = w1B + (ct * 16 + lrow) * TS;
#pragma unroll
      for (int vg = 0; vg < 8; ++vg)
        Bf.u[vg] = *reinterpret_cast<const unsigned*>(wrow + ((vg & 4) ? 16 : 0) + half * 8 + 2 * (vg & 3));
      acc[ct].v = __builtin_amdgcn_wmma_f32_16x16x32_bf16(
          false, A.v, false, Bf.v, (short)0, acc[ct].v, false, false);
    }
  }
}

// ---------------------------------------------------------------------------
// Pass 1: per-channel sum / sumsq of h over all (b,i,j) for BN1 batch stats
// ---------------------------------------------------------------------------
__global__ __launch_bounds__(256) void k_h_stats(
    const unsigned short* __restrict__ xg, const unsigned short* __restrict__ w1g,
    float* __restrict__ gsum, float* __restrict__ gsq) {
  __shared__ unsigned short xjS[2 * 128 * TS];
  __shared__ unsigned short w1S[2 * 128 * TS];
  __shared__ unsigned short xiS[256];
  __shared__ float lsum[128], lsq[128];
  const int tid = threadIdx.x;
  const int bi = blockIdx.x, b = bi >> 8, i = bi & 255;
  if (tid < 128) {
    lsum[tid] = 0.f; lsq[tid] = 0.f;
    reinterpret_cast<unsigned*>(xiS)[tid] =
        reinterpret_cast<const unsigned*>(xg + (((size_t)b * NN + i) << 8))[tid];
  }
  const int lane = tid & 31, lrow = lane & 15;
  float s1[8], s2[8];
#pragma unroll
  for (int ct = 0; ct < 8; ++ct) { s1[ct] = 0.f; s2[ct] = 0.f; }
  F32Acc acc[8];
  for (int jb = 0; jb < 2; ++jb) {
    h_gemm(xg, w1g, xjS, w1S, xiS, b, jb * 128, tid, acc);
#pragma unroll
    for (int ct = 0; ct < 8; ++ct)
#pragma unroll
      for (int r = 0; r < 8; ++r) {
        float v = acc[ct].f[r];
        s1[ct] += v; s2[ct] += v * v;
      }
  }
#pragma unroll
  for (int ct = 0; ct < 8; ++ct) {
    float a = s1[ct] + __shfl_down(s1[ct], 16, 32);
    float q = s2[ct] + __shfl_down(s2[ct], 16, 32);
    if (lane < 16) { atomicAdd(&lsum[ct * 16 + lrow], a); atomicAdd(&lsq[ct * 16 + lrow], q); }
  }
  __syncthreads();
  if (tid < 128) { atomicAdd(&gsum[tid], lsum[tid]); atomicAdd(&gsq[tid], lsq[tid]); }
}

__global__ void k_bn1_fin(const float* gsum, const float* gsq, const float* g,
                          const float* bt, float* oa, float* ob) {
  int t = threadIdx.x; if (t >= DMID) return;
  const float cnt = (float)BB * NN * NN;   // 524288
  float mu = gsum[t] / cnt;
  float var = gsq[t] / cnt - mu * mu;
  float a = g[t] * rsqrtf(var + 1e-5f);
  oa[t] = a; ob[t] = bt[t] - mu * a;
}

// ---------------------------------------------------------------------------
// Pass 2: recompute h, BN1 affine + leaky, dot conv2_w -> logits s[j],
// mask by A_init, softmax over j, agg[b,i,:] = sum_j A[j]*x[b,j,:]  (bf16 out)
// ---------------------------------------------------------------------------
__global__ __launch_bounds__(256) void k_attn(
    const unsigned short* __restrict__ xg, const unsigned short* __restrict__ w1g,
    const float* __restrict__ xf, const int* __restrict__ Ainit,
    const float* __restrict__ w2, const float* __restrict__ bn1a,
    const float* __restrict__ bn1b, unsigned short* __restrict__ aggbf) {
  __shared__ unsigned short xjS[2 * 128 * TS];
  __shared__ unsigned short w1S[2 * 128 * TS];
  __shared__ unsigned short xiS[256];
  __shared__ float w2S[128], aS[128], bS[128];
  __shared__ float sS[256], redS[256];
  const int tid = threadIdx.x;
  const int bi = blockIdx.x, b = bi >> 8, i = bi & 255;
  if (tid < 128) {
    reinterpret_cast<unsigned*>(xiS)[tid] =
        reinterpret_cast<const unsigned*>(xg + (((size_t)b * NN + i) << 8))[tid];
    w2S[tid] = w2[tid]; aS[tid] = bn1a[tid]; bS[tid] = bn1b[tid];
  }
  const int lane = tid & 31, wave = tid >> 5, lrow = lane & 15, half = lane >> 4;
  F32Acc acc[8];
  for (int jb = 0; jb < 2; ++jb) {
    h_gemm(xg, w1g, xjS, w1S, xiS, b, jb * 128, tid, acc);
    float ps[8];
#pragma unroll
    for (int r = 0; r < 8; ++r) ps[r] = 0.f;
#pragma unroll
    for (int ct = 0; ct < 8; ++ct) {
      int c = ct * 16 + lrow;
      float ac = aS[c], bc = bS[c], wc = w2S[c];
#pragma unroll
      for (int r = 0; r < 8; ++r) {
        float hn = ac * acc[ct].f[r] + bc;
        hn = hn >= 0.f ? hn : 0.01f * hn;      // leaky_relu
        ps[r] += wc * hn;
      }
    }
#pragma unroll
    for (int r = 0; r < 8; ++r) {              // reduce 16 lanes -> one column sum
      float v = ps[r];
      v += __shfl_xor(v, 1, 32); v += __shfl_xor(v, 2, 32);
      v += __shfl_xor(v, 4, 32); v += __shfl_xor(v, 8, 32);
      if (lrow == 0) sS[jb * 128 + wave * 16 + half * 8 + r] = v;
    }
  }
  __syncthreads();
  // masked softmax over j
  float sv = sS[tid];
  int m = Ainit[(((size_t)b * NN + i) << 8) + tid];
  sv = (m > 0) ? sv : -9.0e15f;
  redS[tid] = sv; __syncthreads();
  for (int off = 128; off > 0; off >>= 1) {
    if (tid < off) redS[tid] = fmaxf(redS[tid], redS[tid + off]);
    __syncthreads();
  }
  float mx = redS[0]; __syncthreads();
  float e = expf(sv - mx);
  redS[tid] = e; __syncthreads();
  for (int off = 128; off > 0; off >>= 1) {
    if (tid < off) redS[tid] += redS[tid + off];
    __syncthreads();
  }
  float p = e / redS[0]; __syncthreads();
  sS[tid] = p; __syncthreads();
  // aggregation: thread = output feature d
  const float* xb = xf + ((size_t)b << 16);
  float a2 = 0.f;
  for (int j = 0; j < 256; ++j) a2 += sS[j] * xb[(size_t)j * DIN + tid];
  aggbf[(((size_t)b * NN + i) << 8) + tid] = f2bf(a2);
}

// ---------------------------------------------------------------------------
// FC GEMM: Y[2048,256] = AGG[2048,256] @ fc_w^T + fc_b  (+ BN2 stats)
// 256 threads = 8 waves = 4 row strips x 2 column halves; WG tile 64x256.
// Same double-buffered async-to-LDS staging as h_gemm.
// ---------------------------------------------------------------------------
__global__ __launch_bounds__(256) void k_fc(
    const unsigned short* __restrict__ aggbf, const unsigned short* __restrict__ fwg,
    const float* __restrict__ fcb, float* __restrict__ yws,
    float* __restrict__ gsum, float* __restrict__ gsq) {
  __shared__ unsigned short aT[2 * 64 * TS];
  __shared__ unsigned short fw[2 * 256 * TS];
  __shared__ float lsum[256], lsq[256];
  const int tid = threadIdx.x;
  const int rowbase = blockIdx.x * 64;
  lsum[tid] = 0.f; lsq[tid] = 0.f;
  const int lane = tid & 31, wave = tid >> 5, lrow = lane & 15, half = lane >> 4;
  const int rs = wave >> 1, ch = wave & 1;
  F32Acc acc[8];
#pragma unroll
  for (int ct = 0; ct < 8; ++ct)
#pragma unroll
    for (int r = 0; r < 8; ++r) acc[ct].f[r] = 0.f;

  const int arow_ld = tid >> 2, aseg = tid & 3;
  const unsigned long long sa =
      (unsigned long long)(size_t)(aggbf + ((size_t)rowbase << 8));
  const unsigned long long sf = (unsigned long long)(size_t)fwg;
  unsigned lds_a[2], lds_f[2];
#pragma unroll
  for (int p = 0; p < 2; ++p) {
    lds_a[p] = (unsigned)(size_t)(aT + p * 64 * TS + arow_ld * TS + aseg * 8);
    lds_f[p] = (unsigned)(size_t)(fw + p * 256 * TS + tid * TS);
  }
  auto issue = [&](int kc) {
    const int p = kc & 1, k0 = kc * 32;
    // agg[rowbase+row, k0..k0+31]: 16 B per lane
    async_b128(lds_a[p], (unsigned)(((arow_ld << 8) + k0 + aseg * 8) * 2), sa);
    // fc_w[c=tid, k0..k0+31]: 64 B per lane
    async_b128x4(lds_f[p], (unsigned)(((tid << 8) + k0) * 2), sf);
  };

  issue(0);
  for (int kc = 0; kc < 8; ++kc) {
    wait_async0();
    __syncthreads();
    if (kc < 7) issue(kc + 1);
    const unsigned short* aB = aT + (kc & 1) * 64 * TS;
    const unsigned short* fB = fw + (kc & 1) * 256 * TS;

    BF16Frag A;
    const unsigned short* arow = aB + (rs * 16 + lrow) * TS;
#pragma unroll
    for (int vg = 0; vg < 8; ++vg) {
      int kl = ((vg & 4) ? 16 : 0) + half * 8 + 2 * (vg & 3);
      A.u[vg] = *reinterpret_cast<const unsigned*>(arow + kl);
    }
#pragma unroll
    for (int ct = 0; ct < 8; ++ct) {
      BF16Frag Bf;
      const unsigned short* wrow = fB + (ch * 128 + ct * 16 + lrow) * TS;
#pragma unroll
      for (int vg = 0; vg < 8; ++vg)
        Bf.u[vg] = *reinterpret_cast<const unsigned*>(wrow + ((vg & 4) ? 16 : 0) + half * 8 + 2 * (vg & 3));
      acc[ct].v = __builtin_amdgcn_wmma_f32_16x16x32_bf16(
          false, A.v, false, Bf.v, (short)0, acc[ct].v, false, false);
    }
  }
  const int mbase = rowbase + rs * 16 + (half ? 8 : 0);
#pragma unroll
  for (int ct = 0; ct < 8; ++ct) {
    int c = ch * 128 + ct * 16 + lrow;
    float fb = fcb[c];
    float s1 = 0.f, s2 = 0.f;
#pragma unroll
    for (int r = 0; r < 8; ++r) {
      float v = acc[ct].f[r] + fb;
      yws[(size_t)(mbase + r) * DOUT + c] = v;
      s1 += v; s2 += v * v;
    }
    s1 += __shfl_xor(s1, 16, 32);
    s2 += __shfl_xor(s2, 16, 32);
    if (lane < 16) { atomicAdd(&lsum[c], s1); atomicAdd(&lsq[c], s2); }
  }
  __syncthreads();
  atomicAdd(&gsum[tid], lsum[tid]);
  atomicAdd(&gsq[tid], lsq[tid]);
}

__global__ void k_bn2_fin(const float* gsum, const float* gsq, const float* g,
                          const float* bt, float* oa, float* ob) {
  int t = threadIdx.x;
  const float cnt = (float)BB * NN;  // 2048
  float mu = gsum[t] / cnt;
  float var = gsq[t] / cnt - mu * mu;
  float a = g[t] * rsqrtf(var + 1e-5f);
  oa[t] = a; ob[t] = bt[t] - mu * a;
}

__global__ void k_bn2_apply(const float* __restrict__ y, const float* __restrict__ a2,
                            const float* __restrict__ b2, float* __restrict__ out) {
  int idx = blockIdx.x * 256 + threadIdx.x;
  int c = idx & 255;
  out[idx] = y[idx] * a2[c] + b2[c];
}

// ---------------------------------------------------------------------------
// Helpers
// ---------------------------------------------------------------------------
__global__ void k_zero(float* p, int n) {
  int i = blockIdx.x * blockDim.x + threadIdx.x;
  if (i < n) p[i] = 0.f;
}
__global__ void k_cvt_bf(const float* __restrict__ s, unsigned short* __restrict__ d, int n) {
  int i = blockIdx.x * blockDim.x + threadIdx.x;
  if (i < n) d[i] = f2bf(s[i]);
}

// ---------------------------------------------------------------------------
// Launch
// ---------------------------------------------------------------------------
extern "C" void kernel_launch(void* const* d_in, const int* in_sizes, int n_in,
                              void* d_out, int out_size, void* d_ws, size_t ws_size,
                              hipStream_t stream) {
  const float* x       = (const float*)d_in[0];
  const int*   Ainit   = (const int*)d_in[1];
  const float* conv1_w = (const float*)d_in[2];
  // d_in[3] conv1_b: cancels through BN1 (mean-subtraction)
  const float* bn1_g   = (const float*)d_in[4];
  const float* bn1_b   = (const float*)d_in[5];
  const float* conv2_w = (const float*)d_in[6];
  // d_in[7] conv2_b: cancels through softmax shift-invariance
  const float* fc_w    = (const float*)d_in[8];
  const float* fc_b    = (const float*)d_in[9];
  const float* bn2_g   = (const float*)d_in[10];
  const float* bn2_b   = (const float*)d_in[11];
  float* out = (float*)d_out;

  char* ws = (char*)d_ws;
  unsigned short* x_bf   = (unsigned short*)(ws + 0);        // 1 MB
  unsigned short* w1_bf  = (unsigned short*)(ws + 1048576);  // 64 KB
  unsigned short* fcw_bf = (unsigned short*)(ws + 1114112);  // 128 KB
  unsigned short* agg_bf = (unsigned short*)(ws + 1245184);  // 1 MB
  float* yws     = (float*)(ws + 2293760);                   // 2 MB
  float* bn1_sum = (float*)(ws + 4390912);
  float* bn1_sq  = (float*)(ws + 4391424);
  float* bn1_a   = (float*)(ws + 4391936);
  float* bn1_bc  = (float*)(ws + 4392448);
  float* bn2_sum = (float*)(ws + 4392960);
  float* bn2_sq  = (float*)(ws + 4393984);
  float* bn2_a   = (float*)(ws + 4395008);
  float* bn2_bc  = (float*)(ws + 4396032);

  // zero stat accumulators (bn1_sum+bn1_sq contiguous; bn2_sum+bn2_sq contiguous)
  k_zero<<<1, 256, 0, stream>>>(bn1_sum, 256);
  k_zero<<<2, 256, 0, stream>>>(bn2_sum, 512);

  // fp32 -> bf16 staging (x, conv1_w, fc_w are already [row][k] layouts we want)
  k_cvt_bf<<<2048, 256, 0, stream>>>(x, x_bf, BB * NN * DIN);
  k_cvt_bf<<<128, 256, 0, stream>>>(conv1_w, w1_bf, DMID * DIN);
  k_cvt_bf<<<256, 256, 0, stream>>>(fc_w, fcw_bf, DOUT * DIN);

  // pass 1: BN1 batch stats of h
  k_h_stats<<<BB * NN, 256, 0, stream>>>(x_bf, w1_bf, bn1_sum, bn1_sq);
  k_bn1_fin<<<1, 128, 0, stream>>>(bn1_sum, bn1_sq, bn1_g, bn1_b, bn1_a, bn1_bc);

  // pass 2: recompute h, BN+leaky+conv2 -> masked softmax -> agg
  k_attn<<<BB * NN, 256, 0, stream>>>(x_bf, w1_bf, x, Ainit, conv2_w,
                                      bn1_a, bn1_bc, agg_bf);

  // FC GEMM + BN2 stats, then finalize and apply
  k_fc<<<(BB * NN) / 64, 256, 0, stream>>>(agg_bf, fcw_bf, fc_b, yws, bn2_sum, bn2_sq);
  k_bn2_fin<<<1, 256, 0, stream>>>(bn2_sum, bn2_sq, bn2_g, bn2_b, bn2_a, bn2_bc);
  k_bn2_apply<<<(BB * NN * DOUT) / 256, 256, 0, stream>>>(yws, bn2_a, bn2_bc, out);
}
